// CACProjector_70274254897292
// MI455X (gfx1250) — matrix operand order repaired and checked by reading, not compiled, not run on hardware
//
#include <hip/hip_runtime.h>
#include <hip/hip_bf16.h>
#include <math.h>

// ---------------------------------------------------------------------------
// CAC projector: logits = x @ W^T ; distances = ||logits_row - alpha*e_j||
//   x: (16384, 768) f32, W: (1024, 768) f32
//   out = [logits (16384x1024) | distances (16384x1024)] f32
//
// GEMM runs as bf16 hi/lo split (Markidis) on v_wmma_f32_16x16x32_bf16:
//   x = hi(x) + lo(x);  acc += hi*hi + hi*lo + lo*hi  (f32 accumulate)
// which recovers ~fp32 accuracy while using the fast bf16 WMMA pipe.
// ---------------------------------------------------------------------------

typedef __attribute__((ext_vector_type(16))) __bf16    v16bf;
typedef __attribute__((ext_vector_type(4)))  __bf16    v4bf;
typedef __attribute__((ext_vector_type(8)))  float     v8f;
typedef __attribute__((ext_vector_type(4)))  float     f32x4;
typedef __attribute__((ext_vector_type(4)))  unsigned  u32x4;

#define B_DIM 16384
#define D_DIM 768
#define C_DIM 1024
#define ALPHA 10.0f

#define BM 128
#define BN 128
#define BK 32
#define LDT 40            // padded bf16 row stride (80 B, 16B-aligned)

// Load a 16-bf16 A/B fragment from two 16-byte LDS chunks (ds_load_b128 x2).
__device__ __forceinline__ v16bf ld_frag(const __bf16* p0, const __bf16* p1) {
    struct S { u32x4 a, b; } s;
    s.a = *(const u32x4*)p0;
    s.b = *(const u32x4*)p1;
    return __builtin_bit_cast(v16bf, s);
}

__global__ __launch_bounds__(256)
void cac_gemm_kernel(const float* __restrict__ x,
                     const float* __restrict__ W,
                     float* __restrict__ logits) {
    __shared__ __bf16 Ah[BM][LDT];
    __shared__ __bf16 Al[BM][LDT];
    __shared__ __bf16 Bh[BN][LDT];
    __shared__ __bf16 Bl[BN][LDT];

    const int tid  = threadIdx.x;
    const int lane = tid & 31;
    const int wid  = tid >> 5;
    const int wm   = wid & 3;     // 4 waves along M  -> wave covers 32 rows
    const int wn   = wid >> 2;    // 2 waves along N  -> wave covers 64 cols
    const int bm   = blockIdx.y * BM;
    const int bn   = blockIdx.x * BN;

    // global tile-load mapping: 256 threads, 4 iters, 32 rows/iter, 8 f32x4/row
    const int lrow = tid >> 3;          // 0..31
    const int lcol = (tid & 7) * 4;     // 0,4,...,28

    v8f acc[2][4];
    #pragma unroll
    for (int mt = 0; mt < 2; ++mt)
        #pragma unroll
        for (int nt = 0; nt < 4; ++nt)
            acc[mt][nt] = (v8f){0.f,0.f,0.f,0.f,0.f,0.f,0.f,0.f};

    // fragment lane decomposition (matches CDNA5 WMMA VGPR layouts)
    const int fr    = lane & 15;
    const int aksel = (lane >> 4) * 8;   // A: lanes0-15 K{0..7,16..23}; 16-31 K{8..15,24..31}
    const int bksel = (lane >> 4) * 8;   // B: lanes0-15 K{0..7,16..23}; 16-31 K{8..15,24..31}

    for (int k0 = 0; k0 < D_DIM; k0 += BK) {
        // ---- stage f32 tiles -> bf16 hi/lo in LDS -------------------------
        #pragma unroll
        for (int it = 0; it < 4; ++it) {
            const int r = it * 32 + lrow;
            f32x4 av = *(const f32x4*)(x + (size_t)(bm + r) * D_DIM + k0 + lcol);
            f32x4 bv = *(const f32x4*)(W + (size_t)(bn + r) * D_DIM + k0 + lcol);
            v4bf ah, al, bh, bl;
            #pragma unroll
            for (int j = 0; j < 4; ++j) {
                __bf16 h = (__bf16)av[j];
                ah[j] = h;
                al[j] = (__bf16)(av[j] - (float)h);
                h = (__bf16)bv[j];
                bh[j] = h;
                bl[j] = (__bf16)(bv[j] - (float)h);
            }
            *(v4bf*)&Ah[r][lcol] = ah;   // ds_store_b64
            *(v4bf*)&Al[r][lcol] = al;
            *(v4bf*)&Bh[r][lcol] = bh;
            *(v4bf*)&Bl[r][lcol] = bl;
        }
        __syncthreads();

        // prefetch next K tile into L2/L0 (global_prefetch_b8)
        if (k0 + BK < D_DIM) {
            __builtin_prefetch(x + (size_t)(bm + lrow) * D_DIM + k0 + BK + lcol, 0, 1);
            __builtin_prefetch(W + (size_t)(bn + lrow) * D_DIM + k0 + BK + lcol, 0, 1);
        }

        // ---- load fragments from LDS --------------------------------------
        v16bf afh[2], afl[2], bfh[4], bfl[4];
        #pragma unroll
        for (int mt = 0; mt < 2; ++mt) {
            const int row = wm * 32 + mt * 16 + fr;
            afh[mt] = ld_frag(&Ah[row][aksel], &Ah[row][aksel + 16]);
            afl[mt] = ld_frag(&Al[row][aksel], &Al[row][aksel + 16]);
        }
        #pragma unroll
        for (int nt = 0; nt < 4; ++nt) {
            const int col = wn * 64 + nt * 16 + fr;
            bfh[nt] = ld_frag(&Bh[col][bksel], &Bh[col][bksel + 16]);
            bfl[nt] = ld_frag(&Bl[col][bksel], &Bl[col][bksel + 16]);
        }

        // ---- 2x4 tiles x 3 compensated products = 24 WMMAs per wave -------
        #pragma unroll
        for (int mt = 0; mt < 2; ++mt) {
            #pragma unroll
            for (int nt = 0; nt < 4; ++nt) {
                acc[mt][nt] = __builtin_amdgcn_wmma_f32_16x16x32_bf16(
                    false, afh[mt], false, bfh[nt], (short)0, acc[mt][nt], false, false);
                acc[mt][nt] = __builtin_amdgcn_wmma_f32_16x16x32_bf16(
                    false, afh[mt], false, bfl[nt], (short)0, acc[mt][nt], false, false);
                acc[mt][nt] = __builtin_amdgcn_wmma_f32_16x16x32_bf16(
                    false, afl[mt], false, bfh[nt], (short)0, acc[mt][nt], false, false);
            }
        }
        __syncthreads();
    }

    // ---- write logits (C/D layout: VGPR r -> M=r (+8 for lanes 16..31)) ---
    const int rsel = (lane >> 4) * 8;
    #pragma unroll
    for (int mt = 0; mt < 2; ++mt) {
        #pragma unroll
        for (int nt = 0; nt < 4; ++nt) {
            const int col = bn + wn * 64 + nt * 16 + fr;
            #pragma unroll
            for (int r = 0; r < 8; ++r) {
                const int row = bm + wm * 32 + mt * 16 + rsel + r;
                logits[(size_t)row * C_DIM + col] = acc[mt][nt][r];
            }
        }
    }
}

// ---------------------------------------------------------------------------
// Pass 2: per-row squared norm + distances.
//   d[b,j] = sqrt(max(||logits_b||^2 - 2*alpha*logits[b,j] + alpha^2, 0))
// One 256-thread block per row; each thread owns one float4 (256*4 = 1024).
// ---------------------------------------------------------------------------
__global__ __launch_bounds__(256)
void cac_dist_kernel(const float* __restrict__ logits,
                     float* __restrict__ dist) {
    const int row  = blockIdx.x;
    const int tid  = threadIdx.x;
    const int lane = tid & 31;
    const int wid  = tid >> 5;

    const float* lp = logits + (size_t)row * C_DIM;
    f32x4 v = *(const f32x4*)(lp + tid * 4);

    float s = v[0] * v[0] + v[1] * v[1] + v[2] * v[2] + v[3] * v[3];
    #pragma unroll
    for (int off = 16; off > 0; off >>= 1)
        s += __shfl_down(s, off, 32);

    __shared__ float red[8];
    __shared__ float tot;
    if (lane == 0) red[wid] = s;
    __syncthreads();
    if (tid == 0) {
        float t = 0.f;
        #pragma unroll
        for (int i = 0; i < 8; ++i) t += red[i];
        tot = t;
    }
    __syncthreads();

    const float sq = tot;
    f32x4 d;
    #pragma unroll
    for (int j = 0; j < 4; ++j) {
        float d2 = sq - (2.0f * ALPHA) * v[j] + ALPHA * ALPHA;
        d[j] = sqrtf(fmaxf(d2, 0.0f));
    }
    *(f32x4*)(dist + (size_t)row * C_DIM + tid * 4) = d;
}

extern "C" void kernel_launch(void* const* d_in, const int* in_sizes, int n_in,
                              void* d_out, int out_size, void* d_ws, size_t ws_size,
                              hipStream_t stream) {
    (void)in_sizes; (void)n_in; (void)out_size; (void)d_ws; (void)ws_size;
    const float* x = (const float*)d_in[0];
    const float* W = (const float*)d_in[1];
    float* out    = (float*)d_out;
    float* logits = out;                                  // (16384, 1024)
    float* dist   = out + (size_t)B_DIM * C_DIM;          // (16384, 1024)

    dim3 grid(C_DIM / BN, B_DIM / BM);                    // (8, 128)
    cac_gemm_kernel<<<grid, 256, 0, stream>>>(x, W, logits);
    cac_dist_kernel<<<B_DIM, 256, 0, stream>>>(logits, dist);
}
